// CrossViewPAnet_80264348827675
// MI455X (gfx1250) — matrix operand order repaired and checked by reading, not compiled, not run on hardware
//
#include <hip/hip_runtime.h>
#include <hip/hip_bf16.h>
#include <stdint.h>

// ---------------- problem constants ----------------
constexpr int cBS = 2, cV = 4, cH = 24, cW = 24;
constexpr int cC = 256, cK = 16, cL = 6, cNH = 8, cDH = 32, cFF = 1024;
constexpr int cHW = cH * cW;                 // 576
constexpr int cN  = cV * cHW;                // 2304 tokens per batch
constexpr int cBN = cBS * cN;                // 4608 tokens total
constexpr int cBNK = cBN * cK;               // 73728 neighbor rows
constexpr float cEPS = 1e-5f;
constexpr float cSCALE = 0.17677669529663687f; // DH^-0.5

typedef __attribute__((ext_vector_type(16))) __bf16 v16bf;
typedef __attribute__((ext_vector_type(8)))  __bf16 v8bf;
typedef __attribute__((ext_vector_type(8)))  float  v8f;

// ---------------- small utility kernels ----------------

__global__ __launch_bounds__(256) void f32_to_bf16_kernel(const float* __restrict__ src,
                                                          __bf16* __restrict__ dst, int n) {
  int i = blockIdx.x * 256 + threadIdx.x;
  if (i < n) dst[i] = (__bf16)src[i];
}

// feature (B*V, C, H, W) -> out (BN, C) fp32, token order (b, v*HW + pix)
__global__ __launch_bounds__(256) void feat_in_kernel(const float* __restrict__ feat,
                                                      float* __restrict__ out) {
  int e = blockIdx.x * 256 + threadIdx.x;
  if (e >= cBN * cC) return;
  int t = e / cC, c = e % cC;
  int b = t / cN, p = t % cN;
  int v = p / cHW, pix = p % cHW;
  out[e] = feat[(((size_t)(b * cV + v) * cC + c) * cHW) + pix];
}

// out (BN, C) -> d_out (B*V, C, H, W) fp32
__global__ __launch_bounds__(256) void out_permute_kernel(const float* __restrict__ out,
                                                          float* __restrict__ dst) {
  int e = blockIdx.x * 256 + threadIdx.x;
  if (e >= cBS * cV * cC * cHW) return;
  int pix = e % cHW;
  int c   = (e / cHW) % cC;
  int bv  = e / (cHW * cC);
  int b = bv / cV, v = bv % cV;
  int t = b * cN + v * cHW + pix;
  dst[e] = out[(size_t)t * cC + c];
}

// ---------------- KNN (K=16 smallest d2, self included) ----------------
__global__ __launch_bounds__(256) void knn_kernel(const float* __restrict__ xyz,
                                                  int* __restrict__ idx_g) {
  __shared__ float cx[256], cy[256], cz[256];
  const int t = blockIdx.x * 256 + threadIdx.x;   // blocks never straddle a batch (2304/256=9)
  const int b = t / cN;
  const int base = b * cN;
  const float px = xyz[t * 3 + 0], py = xyz[t * 3 + 1], pz = xyz[t * 3 + 2];
  float bd[cK];
  int   bi[cK];
#pragma unroll
  for (int q = 0; q < cK; ++q) { bd[q] = 3.0e38f; bi[q] = base; }
  for (int tile = 0; tile < cN; tile += 256) {
    int cidx = base + tile + threadIdx.x;
    cx[threadIdx.x] = xyz[cidx * 3 + 0];
    cy[threadIdx.x] = xyz[cidx * 3 + 1];
    cz[threadIdx.x] = xyz[cidx * 3 + 2];
    __syncthreads();
    for (int j = 0; j < 256; ++j) {
      float dx = px - cx[j], dy = py - cy[j], dz = pz - cz[j];
      float d2 = dx * dx + dy * dy + dz * dz;
      if (d2 < bd[cK - 1]) {
        bd[cK - 1] = d2; bi[cK - 1] = base + tile + j;
#pragma unroll
        for (int q = cK - 2; q >= 0; --q) {
          if (bd[q + 1] < bd[q]) {
            float td = bd[q]; bd[q] = bd[q + 1]; bd[q + 1] = td;
            int   ti = bi[q]; bi[q] = bi[q + 1]; bi[q + 1] = ti;
          }
        }
      }
    }
    __syncthreads();
  }
#pragma unroll
  for (int q = 0; q < cK; ++q) idx_g[t * cK + q] = bi[q];
}

// ---------------- PE MLP ----------------
__global__ __launch_bounds__(256) void qpe_kernel(const float* __restrict__ b1,
                                                  const float* __restrict__ W2,
                                                  const float* __restrict__ b2,
                                                  float* __restrict__ qpe) {
  __shared__ float hid[cC];
  int c = threadIdx.x;
  hid[c] = fmaxf(b1[c], 0.0f);
  __syncthreads();
  float a = b2[c];
  for (int j = 0; j < cC; ++j) a += W2[c * cC + j] * hid[j];
  qpe[c] = a;
}

__global__ __launch_bounds__(256) void pe_hidden_kernel(const float* __restrict__ xyz,
                                                        const int* __restrict__ idx_g,
                                                        const float* __restrict__ W1,
                                                        const float* __restrict__ b1,
                                                        __bf16* __restrict__ Hpe) {
  int r = blockIdx.x, c = threadIdx.x;
  int t = r >> 4;
  int g = idx_g[r];
  float rx = xyz[g * 3 + 0] - xyz[t * 3 + 0];
  float ry = xyz[g * 3 + 1] - xyz[t * 3 + 1];
  float rz = xyz[g * 3 + 2] - xyz[t * 3 + 2];
  float h = fmaxf(rx * W1[c * 3 + 0] + ry * W1[c * 3 + 1] + rz * W1[c * 3 + 2] + b1[c], 0.0f);
  Hpe[(size_t)r * cC + c] = (__bf16)h;
}

// ---------------- LayerNorm ----------------
__global__ __launch_bounds__(256) void ln_kernel(const float* __restrict__ x,
                                                 const float* __restrict__ w,
                                                 const float* __restrict__ b,
                                                 const float* __restrict__ addv,
                                                 __bf16* __restrict__ yb,
                                                 float* __restrict__ yf) {
  __shared__ float red[256];
  int t = blockIdx.x, c = threadIdx.x;
  float v = x[(size_t)t * cC + c];
  red[c] = v;
  __syncthreads();
  for (int s = 128; s > 0; s >>= 1) { if (c < s) red[c] += red[c + s]; __syncthreads(); }
  float mean = red[0] * (1.0f / cC);
  __syncthreads();
  float d = v - mean;
  red[c] = d * d;
  __syncthreads();
  for (int s = 128; s > 0; s >>= 1) { if (c < s) red[c] += red[c + s]; __syncthreads(); }
  float var = red[0] * (1.0f / cC);
  float y = d * rsqrtf(var + cEPS) * w[c] + b[c];
  if (addv) y += addv[c];
  if (yb) yb[(size_t)t * cC + c] = (__bf16)y;
  else    yf[(size_t)t * cC + c] = y;
}

// ---------------- async copy helper (CDNA5 ASYNCcnt path) ----------------
__device__ __forceinline__ void async_b128(unsigned lds_off, const void* gaddr) {
  asm volatile("global_load_async_to_lds_b128 %0, %1, off"
               :: "v"(lds_off), "v"(gaddr) : "memory");
}
__device__ __forceinline__ void wait_async0() {
  asm volatile("s_wait_asynccnt 0x0" ::: "memory");
}

// ---------------- WMMA GEMM: D = A(MxKd, bf16) @ Bw(NxKd, bf16 row-major = B^T) + bias ----------
// Tile 128x128x32, 256 threads (8 waves), each wave computes 32x64 (8 WMMAs / k-step).
// Double-buffered LDS; B tiles (and A tiles in AMODE 0) staged via GLOBAL_LOAD_ASYNC_TO_LDS_B128.
// AMODE 0: A from memory. AMODE 1: A[r][k] = bf16(gsrc[idx_g[r]][k] (+ addpe[r][k])), sync staging.
// EPI 0: store bf16 | 1: store fp32 | 2: fp32 residual accumulate | 3: relu + bf16
template <int AMODE, int EPI>
__global__ __launch_bounds__(256)
void gemm_bf16_wmma(const __bf16* __restrict__ A, const int* __restrict__ idx_g,
                    const float* __restrict__ gsrc, const __bf16* __restrict__ addpe,
                    const __bf16* __restrict__ Bw, const float* __restrict__ bias,
                    float* __restrict__ Cf, __bf16* __restrict__ Cb,
                    int M, int N, int Kd) {
  __shared__ __align__(16) __bf16 As[2][128][40];   // padded rows (80B) keep 16B alignment
  __shared__ __align__(16) __bf16 Bs[2][128][40];

  const int tid    = threadIdx.x;
  const int wave   = tid >> 5, lane = tid & 31;
  const int wave_m = wave >> 1;         // 0..3 -> 32-row strip
  const int wave_n = wave & 1;          // 0..1 -> 64-col strip
  const int lm     = lane & 15;
  const int lh     = lane >> 4;
  const int m0 = blockIdx.x * 128, n0 = blockIdx.y * 128;

  // staging chunk map: 128 rows x 32 bf16 = 512 x 16B chunks; 2 chunks / thread
  const int ch0_row = tid >> 2,        ch0_k8 = (tid & 3) * 8;
  const int ch1_row = (tid + 256) >> 2, ch1_k8 = ((tid + 256) & 3) * 8;

  v8f acc[2][4] = {};

  auto stageB = [&](int buf, int kt) {
    async_b128((unsigned)(uintptr_t)&Bs[buf][ch0_row][ch0_k8],
               Bw + (size_t)(n0 + ch0_row) * Kd + kt + ch0_k8);
    async_b128((unsigned)(uintptr_t)&Bs[buf][ch1_row][ch1_k8],
               Bw + (size_t)(n0 + ch1_row) * Kd + kt + ch1_k8);
  };
  auto stageA = [&](int buf, int kt) {
    if (AMODE == 0) {
      async_b128((unsigned)(uintptr_t)&As[buf][ch0_row][ch0_k8],
                 A + (size_t)(m0 + ch0_row) * Kd + kt + ch0_k8);
      async_b128((unsigned)(uintptr_t)&As[buf][ch1_row][ch1_k8],
                 A + (size_t)(m0 + ch1_row) * Kd + kt + ch1_k8);
    } else {
#pragma unroll
      for (int c = 0; c < 2; ++c) {
        const int row = c ? ch1_row : ch0_row;
        const int k8  = c ? ch1_k8  : ch0_k8;
        const int r = m0 + row;
        const int g = idx_g[r];
        const float* s = gsrc + (size_t)g * Kd + kt + k8;
        if (addpe) {
          const __bf16* pe = addpe + (size_t)r * Kd + kt + k8;
#pragma unroll
          for (int j = 0; j < 8; ++j) As[buf][row][k8 + j] = (__bf16)(s[j] + (float)pe[j]);
        } else {
#pragma unroll
          for (int j = 0; j < 8; ++j) As[buf][row][k8 + j] = (__bf16)s[j];
        }
      }
    }
  };

  int buf = 0;
  stageA(0, 0);
  stageB(0, 0);

  for (int kt = 0; kt < Kd; kt += 32) {
    wait_async0();        // this wave's async chunks for `buf` are in LDS
    __syncthreads();      // everyone's chunks visible; prev reads of buf^1 done
    if (kt + 32 < Kd) {
      stageA(buf ^ 1, kt + 32);
      stageB(buf ^ 1, kt + 32);
    }
    // ---- compute on current buffer ----
    const __bf16 (*Ab)[40] = As[buf];
    const __bf16 (*Bb)[40] = Bs[buf];
    v16bf af[2];
#pragma unroll
    for (int f = 0; f < 2; ++f) {
      const int arow = wave_m * 32 + f * 16 + lm;
      v8bf alo = *(const v8bf*)&Ab[arow][lh * 8];
      v8bf ahi = *(const v8bf*)&Ab[arow][lh * 8 + 16];
      af[f] = __builtin_shufflevector(alo, ahi, 0, 1, 2, 3, 4, 5, 6, 7,
                                                8, 9, 10, 11, 12, 13, 14, 15);
    }
#pragma unroll
    for (int nt = 0; nt < 4; ++nt) {
      const int brow = wave_n * 64 + nt * 16 + lm;
      v8bf blo = *(const v8bf*)&Bb[brow][lh * 8];
      v8bf bhi = *(const v8bf*)&Bb[brow][lh * 8 + 16];
      v16bf bfrag = __builtin_shufflevector(blo, bhi, 0, 1, 2, 3, 4, 5, 6, 7,
                                                      8, 9, 10, 11, 12, 13, 14, 15);
      acc[0][nt] = __builtin_amdgcn_wmma_f32_16x16x32_bf16(
          false, af[0], false, bfrag, (short)0, acc[0][nt], false, false);
      acc[1][nt] = __builtin_amdgcn_wmma_f32_16x16x32_bf16(
          false, af[1], false, bfrag, (short)0, acc[1][nt], false, false);
    }
    buf ^= 1;
  }

  // ---- epilogue: D VGPR layout -> (m, n) ----
#pragma unroll
  for (int f = 0; f < 2; ++f) {
#pragma unroll
    for (int nt = 0; nt < 4; ++nt) {
#pragma unroll
      for (int r = 0; r < 8; ++r) {
        int m = m0 + wave_m * 32 + f * 16 + r + 8 * lh;
        int n = n0 + wave_n * 64 + nt * 16 + lm;
        float val = acc[f][nt][r] + (bias ? bias[n] : 0.0f);
        size_t o = (size_t)m * N + n;
        if      (EPI == 0) Cb[o] = (__bf16)val;
        else if (EPI == 1) Cf[o] = val;
        else if (EPI == 2) Cf[o] += val;
        else               Cb[o] = (__bf16)fmaxf(val, 0.0f);
      }
    }
  }
}

// ---------------- attention (K=16, NH=8, DH=32), one token per block ----------------
__global__ __launch_bounds__(256) void attn_kernel(const float* __restrict__ q,
                                                   const __bf16* __restrict__ kb,
                                                   const __bf16* __restrict__ vb,
                                                   __bf16* __restrict__ ao) {
  __shared__ float qs[cC];
  __shared__ float ks[cK][cC];
  __shared__ float vs[cK][cC];
  __shared__ float sc[cNH][cK];
  const int t = blockIdx.x, tid = threadIdx.x;
  qs[tid] = q[(size_t)t * cC + tid];
  for (int s = 0; s < cK; ++s) {
    ks[s][tid] = (float)kb[((size_t)t * cK + s) * cC + tid];
    vs[s][tid] = (float)vb[((size_t)t * cK + s) * cC + tid];
  }
  __syncthreads();
  if (tid < cNH * cK) {
    int h = tid >> 4, s = tid & 15;
    float a = 0.0f;
#pragma unroll
    for (int d = 0; d < cDH; ++d) a += qs[h * cDH + d] * ks[s][h * cDH + d];
    sc[h][s] = a * cSCALE;
  }
  __syncthreads();
  if (tid < cNH) {
    float mx = -3.0e38f;
    for (int s = 0; s < cK; ++s) mx = fmaxf(mx, sc[tid][s]);
    float sum = 0.0f;
    for (int s = 0; s < cK; ++s) { float e = __expf(sc[tid][s] - mx); sc[tid][s] = e; sum += e; }
    float inv = 1.0f / sum;
    for (int s = 0; s < cK; ++s) sc[tid][s] *= inv;
  }
  __syncthreads();
  {
    int h = tid >> 5, d = tid & 31;
    float a = 0.0f;
#pragma unroll
    for (int s = 0; s < cK; ++s) a += sc[h][s] * vs[s][h * cDH + d];
    ao[(size_t)t * cC + tid] = (__bf16)a;
  }
}

// ---------------- host orchestration ----------------
extern "C" void kernel_launch(void* const* d_in, const int* in_sizes, int n_in,
                              void* d_out, int out_size, void* d_ws, size_t ws_size,
                              hipStream_t stream) {
  (void)in_sizes; (void)n_in; (void)out_size; (void)ws_size;
  const float* feature   = (const float*)d_in[0];
  const float* xyz       = (const float*)d_in[1];
  const float* pe_W1     = (const float*)d_in[2];
  const float* pe_b1     = (const float*)d_in[3];
  const float* pe_W2     = (const float*)d_in[4];
  const float* pe_b2     = (const float*)d_in[5];
  const float* Wqkv      = (const float*)d_in[6];
  const float* bqkv      = (const float*)d_in[7];
  const float* Wo        = (const float*)d_in[8];
  const float* bo        = (const float*)d_in[9];
  const float* ln_attn_w = (const float*)d_in[10];
  const float* ln_attn_b = (const float*)d_in[11];
  const float* W1        = (const float*)d_in[12];
  const float* b1        = (const float*)d_in[13];
  const float* W2        = (const float*)d_in[14];
  const float* b2        = (const float*)d_in[15];
  const float* ln_ffn_w  = (const float*)d_in[16];
  const float* ln_ffn_b  = (const float*)d_in[17];
  const float* ln_out_w  = (const float*)d_in[18];
  const float* ln_out_b  = (const float*)d_in[19];

  char* ws = (char*)d_ws;
  size_t off = 0;
  auto take = [&](size_t bytes) -> char* {
    char* p = ws + off;
    off += (bytes + 255) & ~(size_t)255;
    return p;
  };
  float*  out   = (float*) take((size_t)cBN * cC * 4);
  float*  qbuf  = (float*) take((size_t)cBN * cC * 4);
  __bf16* t2    = (__bf16*)take((size_t)cBN * cC * 2);
  __bf16* ao    = (__bf16*)take((size_t)cBN * cC * 2);
  float*  qpe   = (float*) take((size_t)cC * 4);
  int*    idxg  = (int*)   take((size_t)cBN * cK * 4);
  __bf16* kpe   = (__bf16*)take((size_t)cBNK * cC * 2);
  __bf16* kbuf  = (__bf16*)take((size_t)cBNK * cC * 2);  // PE-hidden aliases this pre-loop
  __bf16* vbuf  = (__bf16*)take((size_t)cBNK * cC * 2);  // FFN-hidden aliases this
  __bf16* WqkvB = (__bf16*)take((size_t)cL * 3 * cC * cC * 2);
  __bf16* WoB   = (__bf16*)take((size_t)cL * cC * cC * 2);
  __bf16* W1B   = (__bf16*)take((size_t)cL * cFF * cC * 2);
  __bf16* W2B   = (__bf16*)take((size_t)cL * cC * cFF * 2);
  __bf16* peW2B = (__bf16*)take((size_t)cC * cC * 2);
  __bf16* Hpe   = kbuf;
  __bf16* hbuf  = vbuf;

  auto cvt = [&](const float* s, __bf16* d, int n) {
    f32_to_bf16_kernel<<<(n + 255) / 256, 256, 0, stream>>>(s, d, n);
  };
  cvt(Wqkv,  WqkvB, cL * 3 * cC * cC);
  cvt(Wo,    WoB,   cL * cC * cC);
  cvt(W1,    W1B,   cL * cFF * cC);
  cvt(W2,    W2B,   cL * cC * cFF);
  cvt(pe_W2, peW2B, cC * cC);

  feat_in_kernel<<<(cBN * cC) / 256, 256, 0, stream>>>(feature, out);
  knn_kernel<<<cBN / 256, 256, 0, stream>>>(xyz, idxg);
  qpe_kernel<<<1, 256, 0, stream>>>(pe_b1, pe_W2, pe_b2, qpe);
  pe_hidden_kernel<<<cBNK, 256, 0, stream>>>(xyz, idxg, pe_W1, pe_b1, Hpe);
  gemm_bf16_wmma<0, 0><<<dim3(cBNK / 128, cC / 128), 256, 0, stream>>>(
      Hpe, nullptr, nullptr, nullptr, peW2B, pe_b2, nullptr, kpe, cBNK, cC, cC);

  for (int i = 0; i < cL; ++i) {
    const __bf16* Wq = WqkvB + (size_t)i * 3 * cC * cC;
    const __bf16* Wk = Wq + (size_t)cC * cC;
    const __bf16* Wv = Wk + (size_t)cC * cC;
    const float*  bq = bqkv + (size_t)i * 3 * cC;
    const float*  bk = bq + cC;
    const float*  bv = bk + cC;

    ln_kernel<<<cBN, 256, 0, stream>>>(out, ln_attn_w + i * cC, ln_attn_b + i * cC,
                                       qpe, t2, nullptr);
    gemm_bf16_wmma<0, 1><<<dim3(cBN / 128, cC / 128), 256, 0, stream>>>(
        t2, nullptr, nullptr, nullptr, Wq, bq, qbuf, nullptr, cBN, cC, cC);

    gemm_bf16_wmma<1, 0><<<dim3(cBNK / 128, cC / 128), 256, 0, stream>>>(
        nullptr, idxg, out, kpe, Wk, bk, nullptr, kbuf, cBNK, cC, cC);
    gemm_bf16_wmma<1, 0><<<dim3(cBNK / 128, cC / 128), 256, 0, stream>>>(
        nullptr, idxg, out, nullptr, Wv, bv, nullptr, vbuf, cBNK, cC, cC);

    attn_kernel<<<cBN, 256, 0, stream>>>(qbuf, kbuf, vbuf, ao);

    gemm_bf16_wmma<0, 2><<<dim3(cBN / 128, cC / 128), 256, 0, stream>>>(
        ao, nullptr, nullptr, nullptr, WoB + (size_t)i * cC * cC, bo + i * cC,
        out, nullptr, cBN, cC, cC);

    ln_kernel<<<cBN, 256, 0, stream>>>(out, ln_ffn_w + i * cC, ln_ffn_b + i * cC,
                                       nullptr, t2, nullptr);
    gemm_bf16_wmma<0, 3><<<dim3(cBN / 128, cFF / 128), 256, 0, stream>>>(
        t2, nullptr, nullptr, nullptr, W1B + (size_t)i * cFF * cC, b1 + i * cFF,
        nullptr, hbuf, cBN, cFF, cC);
    gemm_bf16_wmma<0, 2><<<dim3(cBN / 128, cC / 128), 256, 0, stream>>>(
        hbuf, nullptr, nullptr, nullptr, W2B + (size_t)i * cC * cFF, b2 + i * cC,
        out, nullptr, cBN, cC, cFF);

    ln_kernel<<<cBN, 256, 0, stream>>>(out, ln_out_w + i * cC, ln_out_b + i * cC,
                                       nullptr, nullptr, out);
  }

  out_permute_kernel<<<(cBS * cV * cC * cHW) / 256, 256, 0, stream>>>(out, (float*)d_out);
}
